// SelfAttention_7756710936661
// MI455X (gfx1250) — compile-verified, matrix operand order checked by reading, not run on hardware
//
#include <hip/hip_runtime.h>

// ---------------------------------------------------------------------------
// CDNA5 (gfx1250) self-attention: fp32 inputs -> bf16 WMMA pipeline
//   K1: Q/K/V = x@W + b   (bf16; V stored transposed Vt[b][d][s])
//   K2: S     = Q@K^T     (bf16, async-LDS double-buffered)
//   K3: P     = softmax(S/32) rows (in place)
//   K4: O     = P@Vt      (f32 out, async-LDS double-buffered)
// ---------------------------------------------------------------------------

typedef __attribute__((ext_vector_type(16))) __bf16 v16bf;
typedef __attribute__((ext_vector_type(8)))  __bf16 v8bf;
typedef __attribute__((ext_vector_type(8)))  float  v8f;
typedef __attribute__((ext_vector_type(4)))  float  v4f;
typedef __attribute__((ext_vector_type(4)))  unsigned int v4u;
typedef __attribute__((ext_vector_type(4)))  int v4i;

// Address-space-qualified int4 pointers for the gfx1250 async-LDS builtins.
typedef __attribute__((address_space(1))) v4i gas_v4i;   // global
typedef __attribute__((address_space(3))) v4i las_v4i;   // LDS

#define BM 128
#define BN 128
#define BK 32
#define LDP 40   // padded LDS row stride in bf16 elems (80 bytes, 16B-aligned)

#ifndef __has_builtin
#define __has_builtin(x) 0
#endif
#if __has_builtin(__builtin_amdgcn_global_load_async_to_lds_b128) && \
    __has_builtin(__builtin_amdgcn_s_wait_asynccnt)
#define HAS_ASYNC 1
#else
#define HAS_ASYNC 0
#endif

__device__ __forceinline__ unsigned short f32_bf16(float f) {
  unsigned int u = __float_as_uint(f);
  u += 0x7FFFu + ((u >> 16) & 1u);           // round-to-nearest-even
  return (unsigned short)(u >> 16);
}
__device__ __forceinline__ float bf16_f32(unsigned short h) {
  return __uint_as_float((unsigned int)h << 16);
}

// 16-byte tile-copy global->LDS. On gfx1250 toolchains with the async
// builtins this is a VGPR-free DMA tracked by ASYNCcnt.
__device__ __forceinline__ void copy16(unsigned short* lds, const unsigned short* g) {
#if HAS_ASYNC
  __builtin_amdgcn_global_load_async_to_lds_b128(
      (gas_v4i*)(unsigned short*)g,
      (las_v4i*)lds,
      /*offset=*/0, /*cpol=*/0);
#else
  *(v4u*)lds = *(const v4u*)g;
#endif
}
__device__ __forceinline__ void async_wait_depth4() {
#if HAS_ASYNC
  __builtin_amdgcn_s_wait_asynccnt(4);   // previous tile's 4 ops done
#endif
}
__device__ __forceinline__ void async_wait_zero() {
#if HAS_ASYNC
  __builtin_amdgcn_s_wait_asynccnt(0);
#endif
}

// Stage a 128x32 bf16 tile (row-major, rowStride elems) into LDS (stride LDP).
// 512 16B chunks / 256 threads = 2 per thread.
__device__ __forceinline__ void stageTile(unsigned short* dst, const unsigned short* src,
                                          size_t rowStride, int tid) {
#pragma unroll
  for (int it = 0; it < 2; ++it) {
    int idx = it * 256 + tid;
    int r = idx >> 2;              // 0..127
    int c = (idx & 3) * 8;         // 0..24
    copy16(dst + r * LDP + c, src + (size_t)r * rowStride + c);
  }
}

// Load a 16x32 bf16 fragment from an LDS tile stored as [row][k], stride LDP.
// CDNA5 16-bit operand layout: lanes 0-15 -> rows 0-15 with K {0..7,16..23};
// lanes 16-31 -> K {8..15,24..31}.
__device__ __forceinline__ v16bf ldsFrag(const unsigned short* base, int row0, int lane) {
  int row = row0 + (lane & 15);
  int kc  = (lane & 16) ? 8 : 0;
  const unsigned short* p = base + row * LDP + kc;
  union { v16bf v; v8bf h[2]; } u;
  u.h[0] = *(const v8bf*)p;          // ds_load_b128
  u.h[1] = *(const v8bf*)(p + 16);   // ds_load_b128
  return u.v;
}

__device__ __forceinline__ v8f wmma_bf16(v16bf a, v16bf b, v8f c) {
  return __builtin_amdgcn_wmma_f32_16x16x32_bf16(
      /*neg_a=*/false, a, /*neg_b=*/false, b,
      /*c_mod=*/(short)0, c, /*reuse_a=*/false, /*reuse_b=*/false);
}

// One K-step of the 128x128 block: wave computes 32x64 = 2x4 fragments.
__device__ __forceinline__ void computeTile(const unsigned short* As_,
                                            const unsigned short* Bs_,
                                            int wm, int wn, int lane,
                                            v8f (&acc)[2][4]) {
  v16bf a[2], b[4];
#pragma unroll
  for (int i = 0; i < 2; ++i) a[i] = ldsFrag(As_, wm + i * 16, lane);
#pragma unroll
  for (int j = 0; j < 4; ++j) b[j] = ldsFrag(Bs_, wn + j * 16, lane);
#pragma unroll
  for (int i = 0; i < 2; ++i)
#pragma unroll
    for (int j = 0; j < 4; ++j)
      acc[i][j] = wmma_bf16(a[i], b[j], acc[i][j]);
}

// ---------------------------------------------------------------------------
// K1: QKV projection GEMM.  M=B*S=16384, N=1024, K=1024.  z selects Q/K/V.
// fp32 operands converted to bf16 during staging.  V written transposed.
// ---------------------------------------------------------------------------
__global__ __launch_bounds__(256) void qkv_gemm(
    const float* __restrict__ x,
    const float* __restrict__ Wq, const float* __restrict__ Wk, const float* __restrict__ Wv,
    const float* __restrict__ bq, const float* __restrict__ bk, const float* __restrict__ bv,
    unsigned short* __restrict__ Q, unsigned short* __restrict__ K,
    unsigned short* __restrict__ Vt)
{
  const int N = 1024, KD = 1024, S = 2048;
  __shared__ __align__(16) unsigned short As[BM * LDP];
  __shared__ __align__(16) unsigned short Bs[BN * LDP];

  const int z = blockIdx.z;
  const float* W    = (z == 0) ? Wq : (z == 1) ? Wk : Wv;
  const float* bias = (z == 0) ? bq : (z == 1) ? bk : bv;
  unsigned short* O = (z == 0) ? Q  : (z == 1) ? K  : Vt;

  const int m0 = blockIdx.x * BM;
  const int n0 = blockIdx.y * BN;
  const int tid  = threadIdx.x;
  const int wave = tid >> 5;
  const int lane = tid & 31;
  const int wm = (wave & 3) * 32;   // 4 waves tile M
  const int wn = (wave >> 2) * 64;  // 2 waves tile N

  v8f acc[2][4] = {};

  for (int k0 = 0; k0 < KD; k0 += BK) {
    // Stage A: x[m0..m0+127][k0..k0+31] fp32 -> bf16 LDS
#pragma unroll
    for (int it = 0; it < 4; ++it) {
      int idx = it * 256 + tid;                 // 0..1023
      int r = idx >> 3;                          // 0..127
      int c = (idx & 7) * 4;                     // 0..28
      v4f v = *(const v4f*)(x + (size_t)(m0 + r) * KD + k0 + c);
      unsigned long long pk =
          (unsigned long long)f32_bf16(v[0]) |
          ((unsigned long long)f32_bf16(v[1]) << 16) |
          ((unsigned long long)f32_bf16(v[2]) << 32) |
          ((unsigned long long)f32_bf16(v[3]) << 48);
      *(unsigned long long*)(As + r * LDP + c) = pk;
    }
    // Stage B (transposed): W[k0+kr][n0+cn] -> Bs[cn][kr]
#pragma unroll
    for (int it = 0; it < 4; ++it) {
      int idx = it * 256 + tid;                 // 0..1023
      int kr = idx >> 5;                         // 0..31
      int cn = (idx & 31) * 4;                   // 0..124
      v4f v = *(const v4f*)(W + (size_t)(k0 + kr) * N + n0 + cn);
#pragma unroll
      for (int i = 0; i < 4; ++i)
        Bs[(cn + i) * LDP + kr] = f32_bf16(v[i]);
    }
    __syncthreads();
    computeTile(As, Bs, wm, wn, lane, acc);
    __syncthreads();
  }

  // Epilogue. Acc layout: VGPR r -> M = i*16 + (lane<16?0:8) + r, N = lane&15.
  const int col   = lane & 15;
  const int rbase = (lane & 16) ? 8 : 0;
  if (z != 2) {
    // Q,K row-major: out[m][n]
#pragma unroll
    for (int j = 0; j < 4; ++j) {
      int n = n0 + wn + j * 16 + col;
      float bsc = bias[n];
#pragma unroll
      for (int i = 0; i < 2; ++i) {
        int mr = m0 + wm + i * 16 + rbase;
#pragma unroll
        for (int r = 0; r < 8; ++r)
          O[(size_t)(mr + r) * N + n] = f32_bf16(acc[i][j][r] + bsc);
      }
    }
  } else {
    // V transposed: Vt[b][n][s]; the 8 acc rows are contiguous in s -> b128 store
    int bb    = m0 >> 11;             // batch (blocks never straddle: 2048 % 128 == 0)
    int sBase = (m0 & (S - 1)) + wm;
#pragma unroll
    for (int j = 0; j < 4; ++j) {
      int n = n0 + wn + j * 16 + col;
      float bsc = bias[n];
#pragma unroll
      for (int i = 0; i < 2; ++i) {
        int s = sBase + i * 16 + rbase;
        union { v4u u; unsigned short h[8]; } t;
#pragma unroll
        for (int r = 0; r < 8; ++r) t.h[r] = f32_bf16(acc[i][j][r] + bsc);
        *(v4u*)(O + ((size_t)bb * N + n) * S + s) = t.u;
      }
    }
  }
}

// ---------------------------------------------------------------------------
// K2: scores = Q @ K^T per batch.  M=N=S=2048, K=D=1024.
// Both operands row-major over d -> pure 16B copies; async + double buffer.
// ---------------------------------------------------------------------------
__global__ __launch_bounds__(256) void scores_gemm(
    const unsigned short* __restrict__ Q, const unsigned short* __restrict__ K,
    unsigned short* __restrict__ Sc,
    long long qkBatchStride, long long scBatchStride)
{
  const int D = 1024, S = 2048;
  __shared__ __align__(16) unsigned short As[2][BM * LDP];
  __shared__ __align__(16) unsigned short Bs[2][BN * LDP];

  const unsigned short* Qb = Q  + (size_t)blockIdx.z * qkBatchStride;
  const unsigned short* Kb = K  + (size_t)blockIdx.z * qkBatchStride;
  unsigned short*       Sb = Sc + (size_t)blockIdx.z * scBatchStride;

  const int m0 = blockIdx.x * BM;
  const int n0 = blockIdx.y * BN;
  const int tid  = threadIdx.x;
  const int wave = tid >> 5;
  const int lane = tid & 31;
  const int wm = (wave & 3) * 32;
  const int wn = (wave >> 2) * 64;

  const unsigned short* Asrc = Qb + (size_t)m0 * D;
  const unsigned short* Bsrc = Kb + (size_t)n0 * D;
  const int T = D / BK;   // 32 K-steps

  v8f acc[2][4] = {};

  stageTile(As[0], Asrc, D, tid);
  stageTile(Bs[0], Bsrc, D, tid);
  for (int k = 0; k < T; ++k) {
    int cur = k & 1;
    if (k + 1 < T) {
      stageTile(As[cur ^ 1], Asrc + (size_t)(k + 1) * BK, D, tid);
      stageTile(Bs[cur ^ 1], Bsrc + (size_t)(k + 1) * BK, D, tid);
      async_wait_depth4();            // wait for tile k only (in-order retire)
    } else {
      async_wait_zero();
    }
    __syncthreads();
    computeTile(As[cur], Bs[cur], wm, wn, lane, acc);
    __syncthreads();                  // readers done before tile k+2 overwrites
  }

  const int col   = lane & 15;
  const int rbase = (lane & 16) ? 8 : 0;
#pragma unroll
  for (int j = 0; j < 4; ++j) {
    int n = n0 + wn + j * 16 + col;
#pragma unroll
    for (int i = 0; i < 2; ++i) {
      int mr = m0 + wm + i * 16 + rbase;
#pragma unroll
      for (int r = 0; r < 8; ++r)
        Sb[(size_t)(mr + r) * S + n] = f32_bf16(acc[i][j][r]);
    }
  }
}

// ---------------------------------------------------------------------------
// K3: in-place row softmax of scores/32.  One wave per 2048-elem row.
// ---------------------------------------------------------------------------
__global__ __launch_bounds__(256) void softmax_rows(
    unsigned short* __restrict__ Sc, long long scBatchStride)
{
  const int S = 2048;
  unsigned short* Sb = Sc + (size_t)blockIdx.z * scBatchStride;
  const int wave = threadIdx.x >> 5;
  const int lane = threadIdx.x & 31;
  const int row  = blockIdx.x * 8 + wave;
  unsigned short* rp = Sb + (size_t)row * S;
  const float scale = 0.03125f;   // 1/sqrt(1024)

  float v[64];
  float m = -3.0e38f;
#pragma unroll
  for (int c = 0; c < 8; ++c) {
    union { v4u u; unsigned short s[8]; } t;
    t.u = *(const v4u*)(rp + c * 256 + lane * 8);
#pragma unroll
    for (int i = 0; i < 8; ++i) {
      float f = bf16_f32(t.s[i]) * scale;
      v[c * 8 + i] = f;
      m = fmaxf(m, f);
    }
  }
#pragma unroll
  for (int off = 16; off > 0; off >>= 1)
    m = fmaxf(m, __shfl_xor(m, off, 32));

  float sum = 0.0f;
#pragma unroll
  for (int i = 0; i < 64; ++i) { v[i] = __expf(v[i] - m); sum += v[i]; }
#pragma unroll
  for (int off = 16; off > 0; off >>= 1)
    sum += __shfl_xor(sum, off, 32);
  float inv = 1.0f / sum;

#pragma unroll
  for (int c = 0; c < 8; ++c) {
    union { v4u u; unsigned short s[8]; } t;
#pragma unroll
    for (int i = 0; i < 8; ++i) t.s[i] = f32_bf16(v[c * 8 + i] * inv);
    *(v4u*)(rp + c * 256 + lane * 8) = t.u;
  }
}

// ---------------------------------------------------------------------------
// K4: O = P @ V per batch.  M=S=2048, N=D=1024, K=S=2048, f32 output.
// A = P row-major; B = Vt[d][s] row-major over s -> pure copies; async + DB.
// ---------------------------------------------------------------------------
__global__ __launch_bounds__(256) void pv_gemm(
    const unsigned short* __restrict__ P, const unsigned short* __restrict__ Vt,
    float* __restrict__ O,
    long long pBatchStride, long long vBatchStride, long long oBatchStride)
{
  const int S = 2048, D = 1024;
  __shared__ __align__(16) unsigned short As[2][BM * LDP];
  __shared__ __align__(16) unsigned short Bs[2][BN * LDP];

  const unsigned short* Pb  = P  + (size_t)blockIdx.z * pBatchStride;
  const unsigned short* Vtb = Vt + (size_t)blockIdx.z * vBatchStride;
  float*                Ob  = O  + (size_t)blockIdx.z * oBatchStride;

  const int m0 = blockIdx.x * BM;   // q rows
  const int n0 = blockIdx.y * BN;   // d cols
  const int tid  = threadIdx.x;
  const int wave = tid >> 5;
  const int lane = tid & 31;
  const int wm = (wave & 3) * 32;
  const int wn = (wave >> 2) * 64;

  const unsigned short* Asrc = Pb  + (size_t)m0 * S;   // stride S over keys
  const unsigned short* Bsrc = Vtb + (size_t)n0 * S;   // Vt rows = d, stride S over keys
  const int T = S / BK;   // 64 K-steps

  v8f acc[2][4] = {};

  stageTile(As[0], Asrc, S, tid);
  stageTile(Bs[0], Bsrc, S, tid);
  for (int k = 0; k < T; ++k) {
    int cur = k & 1;
    if (k + 1 < T) {
      stageTile(As[cur ^ 1], Asrc + (size_t)(k + 1) * BK, S, tid);
      stageTile(Bs[cur ^ 1], Bsrc + (size_t)(k + 1) * BK, S, tid);
      async_wait_depth4();
    } else {
      async_wait_zero();
    }
    __syncthreads();
    computeTile(As[cur], Bs[cur], wm, wn, lane, acc);
    __syncthreads();
  }

  const int col   = lane & 15;
  const int rbase = (lane & 16) ? 8 : 0;
#pragma unroll
  for (int j = 0; j < 4; ++j) {
    int n = n0 + wn + j * 16 + col;
#pragma unroll
    for (int i = 0; i < 2; ++i) {
      int mr = m0 + wm + i * 16 + rbase;
#pragma unroll
      for (int r = 0; r < 8; ++r)
        Ob[(size_t)(mr + r) * D + n] = acc[i][j][r];
    }
  }
}

// ---------------------------------------------------------------------------
extern "C" void kernel_launch(void* const* d_in, const int* in_sizes, int n_in,
                              void* d_out, int out_size, void* d_ws, size_t ws_size,
                              hipStream_t stream)
{
  (void)in_sizes; (void)n_in; (void)out_size;
  const float* x  = (const float*)d_in[0];
  const float* Wq = (const float*)d_in[1];
  const float* bq = (const float*)d_in[2];
  const float* Wk = (const float*)d_in[3];
  const float* bk = (const float*)d_in[4];
  const float* Wv = (const float*)d_in[5];
  const float* bv = (const float*)d_in[6];
  float* out = (float*)d_out;

  const int B = 8, S = 2048, D = 1024;
  const size_t perMat = (size_t)B * S * D;          // bf16 elems per Q/K/Vt
  unsigned short* Q  = (unsigned short*)d_ws;
  unsigned short* K  = Q + perMat;
  unsigned short* Vt = K + perMat;                  // Vt[b][d][s]
  unsigned short* Sc = Vt + perMat;
  const size_t needFull = perMat * 3 * sizeof(unsigned short)
                        + (size_t)B * S * S * sizeof(unsigned short);

  dim3 gQKV((B * S) / BM, D / BN, 3);
  qkv_gemm<<<gQKV, 256, 0, stream>>>(x, Wq, Wk, Wv, bq, bk, bv, Q, K, Vt);

  if (ws_size >= needFull) {
    dim3 gS(S / BM, S / BN, B);
    scores_gemm<<<gS, 256, 0, stream>>>(Q, K, Sc, (long long)S * D, (long long)S * S);
    dim3 gSm(S / 8, 1, B);
    softmax_rows<<<gSm, 256, 0, stream>>>(Sc, (long long)S * S);
    dim3 gPV(S / BM, D / BN, B);
    pv_gemm<<<gPV, 256, 0, stream>>>(Sc, Vt, out,
                                     (long long)S * S, (long long)D * S, (long long)S * D);
  } else {
    // Small-workspace fallback: reuse one S*S bf16 score buffer per batch.
    for (int b = 0; b < B; ++b) {
      dim3 gS(S / BM, S / BN, 1);
      scores_gemm<<<gS, 256, 0, stream>>>(Q + (size_t)b * S * D, K + (size_t)b * S * D,
                                          Sc, 0, 0);
      dim3 gSm(S / 8, 1, 1);
      softmax_rows<<<gSm, 256, 0, stream>>>(Sc, 0);
      dim3 gPV(S / BM, D / BN, 1);
      pv_gemm<<<gPV, 256, 0, stream>>>(Sc, Vt + (size_t)b * D * S, out + (size_t)b * S * D,
                                       0, 0, 0);
    }
  }
}
